// GCNModel_73169062855340
// MI455X (gfx1250) — matrix operand order repaired and checked by reading, not compiled, not run on hardware
//
#include <hip/hip_runtime.h>
#include <hip/hip_bf16.h>

// ---------------------------------------------------------------------------
// 2-layer GCN on MI455X (gfx1250, wave32).
//   h1   = x @ W1                      (f16 WMMA, f32 accum, stored f16)
//   agg1 = Â h1   (Â = D^-1/2 (A+I) D^-1/2, self-loops as non-atomic pre-pass)
//   s    = relu(agg1 + b1) @ W2        (per-node 128-dot, wave reduce)
//   out  = Â s + b2                    (scalar edge scatter)
// ---------------------------------------------------------------------------

#define GCN_N 100000
#define GCN_E 1600000
#define GCN_D 128

typedef __attribute__((ext_vector_type(16))) _Float16 v16h;
typedef __attribute__((ext_vector_type(8)))  _Float16 v8h;
typedef __attribute__((ext_vector_type(8)))  float    v8f;

// Guaranteed-native f32 atomic add (global_atomic_add_f32, no CAS loop).
__device__ __forceinline__ void atomAddF32(float* p, float v) {
    unsafeAtomicAdd(p, v);
}

// ------------------------- init / degree / norm ----------------------------

__global__ void k_deg_init(float* __restrict__ deg) {
    int n = blockIdx.x * blockDim.x + threadIdx.x;
    if (n < GCN_N) deg[n] = 1.0f;                 // self-loop contribution
}

__global__ void k_deg_count(const int* __restrict__ ei, float* __restrict__ deg) {
    int e = blockIdx.x * blockDim.x + threadIdx.x;
    if (e < GCN_E) atomAddF32(&deg[ei[GCN_E + e]], 1.0f);  // exact: counts < 2^24
}

__global__ void k_rsqrt(float* __restrict__ deg_dis) {
    int n = blockIdx.x * blockDim.x + threadIdx.x;
    if (n < GCN_N) {
        float d = deg_dis[n];
        deg_dis[n] = (d > 0.0f) ? rsqrtf(d) : 0.0f;
    }
}

// ------------------------------ f16 converts -------------------------------

__global__ void k_cvt_f16(const float* __restrict__ src, _Float16* __restrict__ dst, int n) {
    int i = blockIdx.x * blockDim.x + threadIdx.x;
    if (i < n) dst[i] = (_Float16)src[i];
}

// ------------------------------ WMMA GEMM ----------------------------------
// h1[N,128] = xh[N,128] @ W1h[128,128]; 256 thr = 8 waves; wave w owns N-tile w.
// 625 blocks x 10 M-tiles each = 6250 M-tiles = 100000 rows exactly.

#define MT_PER_BLOCK 10

__global__ void k_gemm_wmma(const _Float16* __restrict__ xh,
                            const _Float16* __restrict__ W1h,
                            _Float16* __restrict__ h1) {
    const int wave = threadIdx.x >> 5;
    const int lane = threadIdx.x & 31;
    const int r    = lane & 15;        // row/col index within tile
    const int half = lane >> 4;        // lane half selects K sub-range
    const int ncol = wave * 16 + r;    // output column this lane covers (B/C)

    // Preload B fragments for all 4 K-steps (W1h is 32KB, L2-resident).
    v16h bf[4];
#pragma unroll
    for (int kk = 0; kk < 4; ++kk) {
        const int kbase = kk * 32 + half * 8;
#pragma unroll
        for (int i = 0; i < 16; ++i)
            bf[kk][i] = W1h[(size_t)(kbase + i + (i >= 8 ? 8 : 0)) * GCN_D + ncol];
    }

    const int Mbase0 = blockIdx.x * MT_PER_BLOCK * 16;

#pragma unroll 1
    for (int mt = 0; mt < MT_PER_BLOCK; ++mt) {
        const int Mbase = Mbase0 + mt * 16;
        // Single A base pointer: all 8 chunk loads become immediate offsets.
        const v8h* pa = reinterpret_cast<const v8h*>(
            xh + (size_t)(Mbase + r) * GCN_D + half * 8);

        v8f c = {};
#pragma unroll
        for (int kk = 0; kk < 4; ++kk) {
            // A 16x32 f16: lane half 0 -> K {k0+0..7, k0+16..23}; half 1 -> +8
            v8h lo = pa[kk * 4];       // K = k0+kb+0..7
            v8h hi = pa[kk * 4 + 2];   // K = k0+kb+16..23  (+16 halves)
            v16h a;
#pragma unroll
            for (int i = 0; i < 8; ++i) { a[i] = lo[i]; a[8 + i] = hi[i]; }
            c = __builtin_amdgcn_wmma_f32_16x16x32_f16(
                    false, a, false, bf[kk], (short)0, c, false, false);
        }
        // C layout: VGPR v <-> row 8*half + v, col = ncol. One base address,
        // constant 256B row strides fold into the store immediate offset.
        _Float16* op = h1 + (size_t)(Mbase + half * 8) * GCN_D + ncol;
#pragma unroll
        for (int v = 0; v < 8; ++v)
            op[(size_t)v * GCN_D] = (_Float16)c[v];
    }
}

// --------------------- layer-1 aggregation (messages) ----------------------

// Self-loop pre-pass: plain stores fully initialize agg1 (no zeroing needed).
__global__ void k_self_agg(const _Float16* __restrict__ h1,
                           const float* __restrict__ dis,
                           float* __restrict__ agg1) {
    int i = blockIdx.x * blockDim.x + threadIdx.x;
    if (i < GCN_N * GCN_D) {
        int n = i >> 7;
        float d = dis[n];
        agg1[i] = d * d * (float)h1[i];
    }
}

// One wave per edge, 4 features per lane: 8B f16 gather + 4 f32 L2 atomics.
__global__ void k_edge_scatter1(const int* __restrict__ ei,
                                const float* __restrict__ dis,
                                const _Float16* __restrict__ h1,
                                float* __restrict__ agg1) {
    int e    = blockIdx.x * 8 + (threadIdx.x >> 5);
    int lane = threadIdx.x & 31;
    if (e >= GCN_E) return;
    int src = ei[e];
    int dst = ei[GCN_E + e];
    float nrm = dis[src] * dis[dst];

    const _Float16* hp = h1 + (size_t)src * GCN_D + lane * 4;
    float* op = agg1 + (size_t)dst * GCN_D + lane * 4;
#pragma unroll
    for (int j = 0; j < 4; ++j)
        atomAddF32(op + j, nrm * (float)hp[j]);
}

// ------------- layer 2: relu(agg1+b1)@W2, self-loop + bias init ------------

__global__ void k_node2(const float* __restrict__ agg1,
                        const float* __restrict__ b1,
                        const float* __restrict__ W2,
                        const float* __restrict__ b2,
                        const float* __restrict__ dis,
                        float* __restrict__ s,
                        float* __restrict__ out) {
    int n    = blockIdx.x * 8 + (threadIdx.x >> 5);
    int lane = threadIdx.x & 31;
    if (n >= GCN_N) return;
    int f0 = lane * 4;
    const float4 a  = *reinterpret_cast<const float4*>(agg1 + (size_t)n * GCN_D + f0);
    const float4 bb = *reinterpret_cast<const float4*>(b1 + f0);
    const float4 w  = *reinterpret_cast<const float4*>(W2 + f0);
    float acc = fmaxf(a.x + bb.x, 0.0f) * w.x
              + fmaxf(a.y + bb.y, 0.0f) * w.y
              + fmaxf(a.z + bb.z, 0.0f) * w.z
              + fmaxf(a.w + bb.w, 0.0f) * w.w;
#pragma unroll
    for (int off = 16; off > 0; off >>= 1)
        acc += __shfl_xor(acc, off, 32);
    if (lane == 0) {
        float d = dis[n];
        s[n]   = acc;
        out[n] = b2[0] + d * d * acc;   // bias + self-loop message
    }
}

__global__ void k_edge_scatter2(const int* __restrict__ ei,
                                const float* __restrict__ dis,
                                const float* __restrict__ s,
                                float* __restrict__ out) {
    int e = blockIdx.x * blockDim.x + threadIdx.x;
    if (e < GCN_E) {
        int src = ei[e];
        int dst = ei[GCN_E + e];
        atomAddF32(&out[dst], dis[src] * dis[dst] * s[src]);
    }
}

// ------------------------------- launcher ----------------------------------

extern "C" void kernel_launch(void* const* d_in, const int* in_sizes, int n_in,
                              void* d_out, int out_size, void* d_ws, size_t ws_size,
                              hipStream_t stream) {
    const float* x  = (const float*)d_in[0];
    const int*   ei = (const int*)d_in[1];
    const float* W1 = (const float*)d_in[2];
    const float* b1 = (const float*)d_in[3];
    const float* W2 = (const float*)d_in[4];
    const float* b2 = (const float*)d_in[5];
    float* out = (float*)d_out;

    // Workspace carve-up (256B aligned slices).
    char* ws = (char*)d_ws;
    size_t off = 0;
    auto alloc = [&](size_t bytes) {
        void* p = ws + off;
        off = (off + bytes + 255) & ~(size_t)255;
        return p;
    };
    float*    dis  = (float*)   alloc((size_t)GCN_N * 4);           // deg -> dis
    _Float16* xh   = (_Float16*)alloc((size_t)GCN_N * GCN_D * 2);
    _Float16* W1h  = (_Float16*)alloc((size_t)GCN_D * GCN_D * 2);
    _Float16* h1   = (_Float16*)alloc((size_t)GCN_N * GCN_D * 2);
    float*    agg1 = (float*)   alloc((size_t)GCN_N * GCN_D * 4);
    float*    s    = (float*)   alloc((size_t)GCN_N * 4);
    (void)ws_size;

    const int T = 256;
    // Degree / normalization (exact in f32).
    k_deg_init <<<(GCN_N + T - 1) / T, T, 0, stream>>>(dis);
    k_deg_count<<<(GCN_E + T - 1) / T, T, 0, stream>>>(ei, dis);
    k_rsqrt    <<<(GCN_N + T - 1) / T, T, 0, stream>>>(dis);

    // f16 operands for WMMA + halved gather traffic.
    k_cvt_f16<<<(GCN_N * GCN_D + T - 1) / T, T, 0, stream>>>(x,  xh,  GCN_N * GCN_D);
    k_cvt_f16<<<(GCN_D * GCN_D + T - 1) / T, T, 0, stream>>>(W1, W1h, GCN_D * GCN_D);

    // h1 = xh @ W1h via v_wmma_f32_16x16x32_f16.
    k_gemm_wmma<<<(GCN_N / 16) / MT_PER_BLOCK, T, 0, stream>>>(xh, W1h, h1);

    // Layer-1 aggregation.
    k_self_agg     <<<(GCN_N * GCN_D + T - 1) / T, T, 0, stream>>>(h1, dis, agg1);
    k_edge_scatter1<<<(GCN_E + 7) / 8, T, 0, stream>>>(ei, dis, h1, agg1);

    // Layer 2 (fused relu+bias+dot) + scalar edge scatter.
    k_node2        <<<(GCN_N + 7) / 8, T, 0, stream>>>(agg1, b1, W2, b2, dis, s, out);
    k_edge_scatter2<<<(GCN_E + T - 1) / T, T, 0, stream>>>(ei, dis, s, out);
}
